// ProteinFeaturesMembrane_33440615367149
// MI455X (gfx1250) — matrix operand order repaired and checked by reading, not compiled, not run on hardware
//
#include <hip/hip_runtime.h>
#include <hip/hip_bf16.h>

typedef __attribute__((ext_vector_type(16))) _Float16 v16h;
typedef __attribute__((ext_vector_type(8)))  _Float16 v8h;
typedef __attribute__((ext_vector_type(8)))  float    v8f;

#define B_   4
#define L_   1024
#define K_   48
#define EF_  128
#define NF_  128
#define NPE_ 16
#define MAXREL_ 32
#define EIN_ 416                 // 16 + 25*16
#define M_   (B_*L_*K_)          // 196608 edge rows
#define AROW_ 424                // padded LDS row stride (halves); 848B, bank-conflict free

// PAIRS translated to atom ids: N=0, Ca=1, C=2, O=3, Cb=4
__device__ const int PAIR_A[24] = {0,2,3,4,1,1,1,1,0,0,0,4,4,3,0,2,3,4,2,3,4,2,3,2};
__device__ const int PAIR_B[24] = {0,2,3,4,0,2,3,4,2,3,4,2,3,2,1,1,1,1,0,0,0,4,4,3};

__device__ __forceinline__ void get_coord(const float* X, const float* cbw, int bl, int atom,
                                          float& x, float& y, float& z) {
  if (atom < 4) { const float* p = X + (size_t)bl*12 + atom*3; x=p[0]; y=p[1]; z=p[2]; }
  else          { const float* p = cbw + (size_t)bl*3;          x=p[0]; y=p[1]; z=p[2]; }
}

// ---------------- Cb virtual atom ----------------
__global__ void cb_kernel(const float* __restrict__ X, float* __restrict__ cbw) {
  int bl = blockIdx.x*256 + threadIdx.x;
  if (bl >= B_*L_) return;
  const float* p = X + (size_t)bl*12;
  float nx=p[0],ny=p[1],nz=p[2], cax=p[3],cay=p[4],caz=p[5], cx=p[6],cy=p[7],cz=p[8];
  float bx=cax-nx, by=cay-ny, bz=caz-nz;
  float vx=cx-cax, vy=cy-cay, vz=cz-caz;
  float axv=by*vz-bz*vy, ayv=bz*vx-bx*vz, azv=bx*vy-by*vx;
  float* q = cbw + (size_t)bl*3;
  q[0] = -0.58273431f*axv + 0.56802827f*bx - 0.54067466f*vx + cax;
  q[1] = -0.58273431f*ayv + 0.56802827f*by - 0.54067466f*vy + cay;
  q[2] = -0.58273431f*azv + 0.56802827f*bz - 0.54067466f*vz + caz;
}

// ------- pre-swizzle edge_w into per-lane WMMA B-fragment layout (fp16) -------
// layout: wp[((kt*8 + nt)*32 + lane)*16 + h]; lane<16 -> K=kt*32+h, N=nt*16+lane
//                                              lane>=16 -> K=kt*32+16+h, N=nt*16+(lane-16)
__global__ void wprep_kernel(const float* __restrict__ edge_w, _Float16* __restrict__ wp) {
  int idx = blockIdx.x*256 + threadIdx.x;
  if (idx >= 13*8*32*16) return;
  int h    = idx & 15;
  int lane = (idx >> 4) & 31;
  int nt   = (idx >> 9) & 7;
  int kt   = idx >> 12;
  int kk   = kt*32 + ((lane & 16) ? 16 : 0) + h;
  int n    = nt*16 + (lane & 15);
  wp[idx] = (_Float16)edge_w[(size_t)kk*EF_ + n];
}

// ---------------- node embedding + LayerNorm ----------------
__global__ void node_kernel(const int* __restrict__ lab, const float* __restrict__ node_w,
                            const float* __restrict__ g, const float* __restrict__ bb,
                            float* __restrict__ V) {
  __shared__ float r1[128];
  __shared__ float r2[128];
  int row = blockIdx.x, t = threadIdx.x;
  float x = node_w[lab[row]*NF_ + t];
  r1[t] = x; r2[t] = x*x;
  __syncthreads();
  for (int s = 64; s > 0; s >>= 1) {
    if (t < s) { r1[t] += r1[t+s]; r2[t] += r2[t+s]; }
    __syncthreads();
  }
  float mean = r1[0]*(1.f/128.f);
  float var  = r2[0]*(1.f/128.f) - mean*mean;
  V[(size_t)row*NF_ + t] = (x - mean)*rsqrtf(var + 1e-5f)*g[t] + bb[t];
}

// ---------------- top-K=48 neighbors by masked Ca distance ----------------
// key = (f32bits(D_adj) << 32) | j  -> u64 min == smallest D, smallest index tiebreak.
// Candidates live in registers (4/thread); reductions are wave32 shfl_xor trees,
// cross-wave combine via an 8-entry LDS array: 2 barriers per extraction.
__global__ void __launch_bounds__(256) topk_kernel(const float* __restrict__ X,
                                                   const float* __restrict__ mask,
                                                   float* __restrict__ dnb,
                                                   int* __restrict__ eidx,
                                                   float* __restrict__ eidx_f) {
  __shared__ unsigned long long redu[8];
  __shared__ float redf[8];
  __shared__ unsigned long long winner;
  const int row = blockIdx.x;
  const int b = row >> 10, i = row & (L_-1);
  const int t = threadIdx.x;
  const int lane = t & 31, wave = t >> 5;
  const float* cai = X + ((size_t)(b*L_ + i)*4 + 1)*3;
  const float cx=cai[0], cy=cai[1], cz=cai[2];
  const float mi = mask[b*L_ + i];

  float dv[4], m2v[4];
  float lmax = 0.f;
  #pragma unroll
  for (int q = 0; q < 4; ++q) {
    int j = t + q*256;
    const float* cj = X + ((size_t)(b*L_ + j)*4 + 1)*3;
    float dx=cx-cj[0], dy=cy-cj[1], dz=cz-cj[2];
    float m2 = mi * mask[b*L_ + j];
    float d  = m2 * sqrtf(dx*dx + dy*dy + dz*dz + 1e-6f);
    dv[q] = d; m2v[q] = m2;
    lmax = fmaxf(lmax, d);
  }
  #pragma unroll
  for (int off = 16; off >= 1; off >>= 1) lmax = fmaxf(lmax, __shfl_xor(lmax, off, 32));
  if (lane == 0) redf[wave] = lmax;
  __syncthreads();
  float dmax = redf[0];
  #pragma unroll
  for (int w = 1; w < 8; ++w) dmax = fmaxf(dmax, redf[w]);

  unsigned long long kreg[4];
  #pragma unroll
  for (int q = 0; q < 4; ++q) {
    int j = t + q*256;
    float dadj = dv[q] + (1.f - m2v[q])*dmax;
    kreg[q] = ((unsigned long long)__float_as_uint(dadj) << 32) | (unsigned int)j;
  }

  for (int k = 0; k < K_; ++k) {
    unsigned long long best = kreg[0];
    #pragma unroll
    for (int q = 1; q < 4; ++q) if (kreg[q] < best) best = kreg[q];
    #pragma unroll
    for (int off = 16; off >= 1; off >>= 1) {
      unsigned long long o = __shfl_xor(best, off, 32);
      if (o < best) best = o;
    }
    if (lane == 0) redu[wave] = best;
    __syncthreads();
    if (t == 0) {
      unsigned long long w = redu[0];
      #pragma unroll
      for (int q = 1; q < 8; ++q) if (redu[q] < w) w = redu[q];
      winner = w;
      int j = (int)(w & 0xffffffffu);
      int m = row*K_ + k;
      dnb[m]    = __uint_as_float((unsigned int)(w >> 32));
      eidx[m]   = j;
      eidx_f[m] = (float)j;
    }
    __syncthreads();
    unsigned long long w = winner;
    int j = (int)(w & 0xffffffffu);
    if ((j & 255) == t) kreg[j >> 8] = ~0ull;   // owner invalidates the extracted key
  }
}

// ------- fused: feature tile -> LDS -> WMMA GEMM -> in-register LayerNorm -> E -------
// 128 threads (4 waves), 64 edge-rows per block. LDS A tile: 64 x 424 halves (54 KB).
__global__ void __launch_bounds__(128) edge_kernel(
    const float* __restrict__ X, const float* __restrict__ cbw,
    const float* __restrict__ dnb, const int* __restrict__ eidx,
    const int* __restrict__ Ridx, const int* __restrict__ chain,
    const float* __restrict__ pe_w, const float* __restrict__ pe_b,
    const _Float16* __restrict__ wp,
    const float* __restrict__ ne_g, const float* __restrict__ ne_b,
    float* __restrict__ E) {
  extern __shared__ _Float16 As[];          // [64][AROW_]
  const int m0  = blockIdx.x * 64;
  const int tid = threadIdx.x;

  // warm L2 with the B-fragment weight stream (global_prefetch_b8)
  __builtin_prefetch(wp + (size_t)tid * 416, 0, 1);

  // ---- phase 1: build fp16 feature tile in LDS (26 sets of 16 per row) ----
  for (int s = tid; s < 64*26; s += 128) {
    int rl = s / 26, set = s - rl*26;
    int m = m0 + rl;
    int b = m / (L_*K_);
    int rem = m - b*(L_*K_);
    int i = rem / K_;
    int bi = b*L_ + i;
    int j  = eidx[m];
    int bj = b*L_ + j;
    _Float16* dst = As + rl*AROW_ + set*16;
    if (set == 0) {                                    // positional encoding gather
      int off  = Ridx[bi] - Ridx[bj];
      int same = (chain[bi] == chain[bj]);
      int dcl  = off + MAXREL_;
      dcl = dcl < 0 ? 0 : (dcl > 2*MAXREL_ ? 2*MAXREL_ : dcl);
      int didx = same ? dcl : (2*MAXREL_ + 1);
      const float* pw = pe_w + didx*NPE_;
      #pragma unroll
      for (int tt = 0; tt < 16; ++tt) dst[tt] = (_Float16)(pw[tt] + pe_b[tt]);
    } else {                                           // Gaussian RBF of a distance
      float d;
      if (set == 1) d = dnb[m];
      else {
        int p = set - 2;
        float ax,ay,az,bx,by,bz;
        get_coord(X, cbw, bi, PAIR_A[p], ax,ay,az);
        get_coord(X, cbw, bj, PAIR_B[p], bx,by,bz);
        float dx=ax-bx, dy=ay-by, dz=az-bz;
        d = sqrtf(dx*dx + dy*dy + dz*dz + 1e-6f);
      }
      #pragma unroll
      for (int tt = 0; tt < 16; ++tt) {
        float mu = 2.0f + (20.0f/15.0f)*tt;
        float a  = (d - mu) * 0.8f;                    // sigma = 1.25
        dst[tt] = (_Float16)__expf(-a*a);
      }
    }
  }
  __syncthreads();

  // ---- phase 2: per-wave 16(M) x 128(N) WMMA GEMM, K = 13 x 32 ----
  const int wave = tid >> 5;
  const int lane = tid & 31;
  const int r    = lane & 15;
  const int hi   = lane >> 4;                           // half-wave group
  v8f zero = {};
  v8f acc[8];
  #pragma unroll
  for (int nt = 0; nt < 8; ++nt) acc[nt] = zero;

  const _Float16* arow = As + (wave*16 + r)*AROW_;
  #pragma unroll 1
  for (int kt = 0; kt < 13; ++kt) {
    // A fragment: lane<16 -> K {0..15}, lane>=16 -> K {16..31} (rel.)
    int k0 = kt*32 + hi*16;
    v8h a0 = *(const v8h*)(arow + k0);
    v8h a1 = *(const v8h*)(arow + k0 + 8);
    v16h a;
    #pragma unroll
    for (int e = 0; e < 8; ++e) { a[e] = a0[e]; a[e+8] = a1[e]; }
    #pragma unroll
    for (int nt = 0; nt < 8; ++nt) {
      v16h bf = *(const v16h*)(wp + (((kt*8 + nt)*32 + lane) << 4));
      acc[nt] = __builtin_amdgcn_wmma_f32_16x16x32_f16(false, a, false, bf,
                                                       (short)0, acc[nt], false, false);
    }
  }

  // ---- phase 3: LayerNorm over N=128 per row; rows live in 16-lane half-groups ----
  float sum[8], ssq[8];
  #pragma unroll
  for (int rr = 0; rr < 8; ++rr) {
    float s0 = 0.f, s1 = 0.f;
    #pragma unroll
    for (int nt = 0; nt < 8; ++nt) { float v = acc[nt][rr]; s0 += v; s1 += v*v; }
    #pragma unroll
    for (int off = 1; off < 16; off <<= 1) {            // xor stays within 16-lane group
      s0 += __shfl_xor(s0, off, 32);
      s1 += __shfl_xor(s1, off, 32);
    }
    sum[rr] = s0; ssq[rr] = s1;
  }
  float gg[8], bv[8];
  #pragma unroll
  for (int nt = 0; nt < 8; ++nt) { int n = nt*16 + r; gg[nt] = ne_g[n]; bv[nt] = ne_b[n]; }

  #pragma unroll
  for (int rr = 0; rr < 8; ++rr) {
    float mean = sum[rr]*(1.0f/128.0f);
    float var  = ssq[rr]*(1.0f/128.0f) - mean*mean;
    float rstd = rsqrtf(var + 1e-5f);
    int mrow = m0 + wave*16 + rr + hi*8;
    float* out = E + (size_t)mrow*EF_;
    #pragma unroll
    for (int nt = 0; nt < 8; ++nt) {
      int n = nt*16 + r;
      out[n] = (acc[nt][rr] - mean)*rstd*gg[nt] + bv[nt];
    }
  }
}

extern "C" void kernel_launch(void* const* d_in, const int* in_sizes, int n_in,
                              void* d_out, int out_size, void* d_ws, size_t ws_size,
                              hipStream_t stream) {
  (void)in_sizes; (void)n_in; (void)out_size; (void)ws_size;
  const float* X      = (const float*)d_in[0];
  const float* mask   = (const float*)d_in[1];
  const int*   Ridx   = (const int*)  d_in[2];
  const int*   chain  = (const int*)  d_in[3];
  const int*   mlab   = (const int*)  d_in[4];
  const float* pe_w   = (const float*)d_in[5];
  const float* pe_b   = (const float*)d_in[6];
  const float* edge_w = (const float*)d_in[7];
  const float* ne_g   = (const float*)d_in[8];
  const float* ne_b   = (const float*)d_in[9];
  const float* node_w = (const float*)d_in[10];
  const float* nn_g   = (const float*)d_in[11];
  const float* nn_b   = (const float*)d_in[12];

  float* V_out  = (float*)d_out;                       // (B,L,128)
  float* E_out  = V_out + (size_t)B_*L_*NF_;           // (B,L,K,128)
  float* Ei_out = E_out + (size_t)M_*EF_;              // (B,L,K)

  char* ws = (char*)d_ws;
  float*    cbw  = (float*)(ws);                               // 4096*3 f32
  float*    dnb  = (float*)(ws + 49152);                       // 196608 f32
  int*      eidx = (int*)  (ws + 49152 + 786432);              // 196608 i32
  _Float16* wp   = (_Float16*)(ws + 49152 + 786432 + 786432);  // 53248 f16

  cb_kernel  <<<(B_*L_ + 255)/256, 256, 0, stream>>>(X, cbw);
  wprep_kernel<<<(13*8*32*16 + 255)/256, 256, 0, stream>>>(edge_w, wp);
  node_kernel<<<B_*L_, 128, 0, stream>>>(mlab, node_w, nn_g, nn_b, V_out);
  topk_kernel<<<B_*L_, 256, 0, stream>>>(X, mask, dnb, eidx, Ei_out);
  edge_kernel<<<M_/64, 128, 64*AROW_*2, stream>>>(X, cbw, dnb, eidx, Ridx, chain,
                                                  pe_w, pe_b, wp, ne_g, ne_b, E_out);
}
